// LinformerAttention_51127290692162
// MI455X (gfx1250) — matrix operand order repaired and checked by reading, not compiled, not run on hardware
//
#include <hip/hip_runtime.h>

// ---------------------------------------------------------------------------
// Linformer attention for MI455X (gfx1250, wave32, WMMA).
// ~34 GFLOP vs ~260 MB traffic -> compute bound -> bf16 WMMA
// (v_wmma_f32_16x16x32_bf16) with f32 accumulation; softmax in f32 registers.
// Data movement: LDS-staged transposes with packed bf16x2 stores, async
// global->LDS (global_load_async_to_lds_b128) for the K_proj/V_proj tiles.
// ---------------------------------------------------------------------------

typedef __attribute__((ext_vector_type(16))) __bf16        v16bf;
typedef __attribute__((ext_vector_type(8)))  float         v8f;
typedef __attribute__((ext_vector_type(4)))  float         v4f;
typedef __attribute__((ext_vector_type(4)))  unsigned int  v4u;

constexpr int Bb = 4, Ls = 4096, Hh = 16, Cc = 64, Rr = 256;

__device__ __forceinline__ unsigned short f32_to_bf16u(float f) {
  union { __bf16 h; unsigned short u; } cv;
  cv.h = (__bf16)f;
  return cv.u;
}

__device__ __forceinline__ unsigned pack_bf16x2(float lo, float hi) {
  union { __bf16 h[2]; unsigned u; } cv;
  cv.h[0] = (__bf16)lo;
  cv.h[1] = (__bf16)hi;
  return cv.u;
}

union FragBF {
  v16bf v;
  v4u   q[2];
  unsigned short us[16];
};

// A-matrix 16x32 bf16 fragment (ISA 7.12.2): lane<16 holds M=lane, K = kcol+[0..7]
// and kcol+[16..23]; lane>=16 holds M=lane-16, K = kcol+[8..15] and kcol+[24..31].
__device__ __forceinline__ v16bf load_frag_a(const unsigned short* base, int stride,
                                             int row, int kcol) {
  const int lane = threadIdx.x & 31;
  const unsigned short* p = base + (row + (lane & 15)) * stride + kcol + ((lane >> 4) << 3);
  FragBF f;
  f.q[0] = *reinterpret_cast<const v4u*>(p);
  f.q[1] = *reinterpret_cast<const v4u*>(p + 16);
  return f.v;
}

// B-matrix 32x16 bf16 fragment: lane<16 holds column N=lane, K = kcol+[0..7] and kcol+[16..23];
// lane>=16 holds column N=lane-16, K = kcol+[8..15] and kcol+[24..31]. (column-major [n][k] in LDS)
__device__ __forceinline__ v16bf load_frag_b(const unsigned short* base, int stride,
                                             int col, int kcol) {
  const int lane = threadIdx.x & 31;
  const unsigned short* p = base + (col + (lane & 15)) * stride + kcol + ((lane >> 4) << 3);
  FragBF f;
  f.q[0] = *reinterpret_cast<const v4u*>(p);
  f.q[1] = *reinterpret_cast<const v4u*>(p + 16);
  return f.v;
}

// Transpose helper: reads 8 contiguous f32 from two consecutive "l" rows of a
// global matrix and writes 8 packed bf16x2 dwords into a [col][l] LDS tile
// (stride 32 halfwords), so each store carries both l-values of the pair.
__device__ __forceinline__ void load_transpose_2x8(const float* row0, const float* row1,
                                                   unsigned short* dstBase, int col0,
                                                   int lpair) {
  v4f a0 = *reinterpret_cast<const v4f*>(row0);
  v4f a1 = *reinterpret_cast<const v4f*>(row0 + 4);
  v4f b0 = *reinterpret_cast<const v4f*>(row1);
  v4f b1 = *reinterpret_cast<const v4f*>(row1 + 4);
  unsigned short* d = dstBase + col0 * 32 + lpair;
  *reinterpret_cast<unsigned*>(d + 0 * 32) = pack_bf16x2(a0.x, b0.x);
  *reinterpret_cast<unsigned*>(d + 1 * 32) = pack_bf16x2(a0.y, b0.y);
  *reinterpret_cast<unsigned*>(d + 2 * 32) = pack_bf16x2(a0.z, b0.z);
  *reinterpret_cast<unsigned*>(d + 3 * 32) = pack_bf16x2(a0.w, b0.w);
  *reinterpret_cast<unsigned*>(d + 4 * 32) = pack_bf16x2(a1.x, b1.x);
  *reinterpret_cast<unsigned*>(d + 5 * 32) = pack_bf16x2(a1.y, b1.y);
  *reinterpret_cast<unsigned*>(d + 6 * 32) = pack_bf16x2(a1.z, b1.z);
  *reinterpret_cast<unsigned*>(d + 7 * 32) = pack_bf16x2(a1.w, b1.w);
}

// ---------------------------------------------------------------------------
// Kernel 1: k_proj[bh][r][c] = sum_l proj_k[l][r] * k[b,l,h,c]   (bf16 out)
//           v_proj stored transposed as [bh][c][r] for stage-3 B fragments.
// Block: 128 threads (4 waves), one (b,h) and a 64-wide r tile.
// Waves 0/1 -> k_proj (32 r rows each), waves 2/3 -> v_proj.
// ---------------------------------------------------------------------------
__global__ void __launch_bounds__(128) linformer_project_kv(
    const float* __restrict__ kin, const float* __restrict__ vin,
    const float* __restrict__ pk,  const float* __restrict__ pv,
    unsigned short* __restrict__ kp, unsigned short* __restrict__ vp) {
  __shared__ unsigned short sAk[64 * 32];  // proj_k^T chunk: [r][l]
  __shared__ unsigned short sBk[64 * 32];  // k^T chunk:      [c][l]
  __shared__ unsigned short sAv[64 * 32];  // proj_v^T chunk
  __shared__ unsigned short sBv[64 * 32];  // v^T chunk

  const int bh = blockIdx.x >> 2;
  const int r0 = (blockIdx.x & 3) * 64;
  const int b  = bh >> 4;
  const int h  = bh & 15;

  const int t    = threadIdx.x;
  const int wave = t >> 5;
  const int lane = t & 31;
  const bool doV = wave >= 2;
  const int m0   = (wave & 1) * 32;

  const v8f vzero = {};
  v8f acc[2][4];
#pragma unroll
  for (int i = 0; i < 2; ++i)
#pragma unroll
    for (int j = 0; j < 4; ++j) acc[i][j] = vzero;

  const int a2 = (t >> 3) << 1;  // l pair base: 0,2,...,30
  const int c8 = (t & 7) * 8;    // 8-column slice

  for (int kc = 0; kc < Ls / 32; ++kc) {
    const int l0 = kc * 32;
    {  // proj_k -> sAk[r][l]
      const float* s0 = pk + (size_t)(l0 + a2) * Rr + r0 + c8;
      load_transpose_2x8(s0, s0 + Rr, sAk, c8, a2);
    }
    {  // proj_v -> sAv[r][l]
      const float* s0 = pv + (size_t)(l0 + a2) * Rr + r0 + c8;
      load_transpose_2x8(s0, s0 + Rr, sAv, c8, a2);
    }
    {  // k -> sBk[c][l]
      const float* s0 = kin + (((size_t)b * Ls + l0 + a2) * Hh + h) * Cc + c8;
      load_transpose_2x8(s0, s0 + Hh * Cc, sBk, c8, a2);
      __builtin_prefetch(s0 + (size_t)32 * Hh * Cc, 0, 1);
    }
    {  // v -> sBv[c][l]
      const float* s0 = vin + (((size_t)b * Ls + l0 + a2) * Hh + h) * Cc + c8;
      load_transpose_2x8(s0, s0 + Hh * Cc, sBv, c8, a2);
      __builtin_prefetch(s0 + (size_t)32 * Hh * Cc, 0, 1);
    }
    __syncthreads();

    const unsigned short* sA = doV ? sAv : sAk;
    const unsigned short* sB = doV ? sBv : sBk;
#pragma unroll
    for (int mt = 0; mt < 2; ++mt) {
      v16bf a = load_frag_a(sA, 32, m0 + mt * 16, 0);
#pragma unroll
      for (int nt = 0; nt < 4; ++nt) {
        v16bf bm = load_frag_b(sB, 32, nt * 16, 0);
        acc[mt][nt] = __builtin_amdgcn_wmma_f32_16x16x32_bf16(
            false, a, false, bm, (short)0, acc[mt][nt], false, false);
      }
    }
    __syncthreads();
  }

  // Epilogue: D layout -> lane(&15)=N, vgpr e -> M = e + 8*(lane>=16)
  const int nlo = lane & 15;
  const int mhi = (lane >> 4) << 3;
  if (!doV) {
#pragma unroll
    for (int mt = 0; mt < 2; ++mt)
#pragma unroll
      for (int nt = 0; nt < 4; ++nt)
#pragma unroll
        for (int e = 0; e < 8; ++e) {
          const int r = r0 + m0 + mt * 16 + mhi + e;
          const int c = nt * 16 + nlo;
          kp[((size_t)bh * Rr + r) * Cc + c] = f32_to_bf16u(acc[mt][nt][e]);
        }
  } else {
#pragma unroll
    for (int mt = 0; mt < 2; ++mt)
#pragma unroll
      for (int nt = 0; nt < 4; ++nt)
#pragma unroll
        for (int e = 0; e < 8; ++e) {
          const int r = r0 + m0 + mt * 16 + mhi + e;
          const int c = nt * 16 + nlo;
          vp[((size_t)bh * Cc + c) * Rr + r] = f32_to_bf16u(acc[mt][nt][e]);
        }
  }
}

// ---------------------------------------------------------------------------
// Kernel 2: scores = Q K_proj^T * scale ; softmax over R ; out = P V_proj.
// 128 threads (4 waves) per (b,h, 64-row L tile). K_proj/V_proj tiles pulled
// into LDS via gfx1250 async global->LDS DMA; softmax fully in registers via
// wave32 shuffles across the 16-lane halves of the WMMA C/D layout.
// ---------------------------------------------------------------------------
// Dynamic-LDS layout (no static LDS in this kernel -> region starts at 0):
constexpr unsigned SQ_OFF  = 0;                          // [64][64]  bf16
constexpr unsigned SKP_OFF = SQ_OFF + 64 * 64 * 2;       // [256][64] bf16
constexpr unsigned SVP_OFF = SKP_OFF + 256 * 64 * 2;     // [64][256] bf16
constexpr unsigned SP_OFF  = SVP_OFF + 64 * 256 * 2;     // [64][256] bf16
constexpr unsigned SMEM_BYTES = SP_OFF + 64 * 256 * 2;   // 104 KB

__global__ void __launch_bounds__(128) linformer_attn(
    const float* __restrict__ q, const unsigned short* __restrict__ kp,
    const unsigned short* __restrict__ vp, float* __restrict__ out) {
  extern __shared__ unsigned short smem[];
  unsigned short* sQ  = smem;              // A for scores      [l][c]
  unsigned short* sKP = sQ + 64 * 64;      // B for scores      [r][c]
  unsigned short* sVP = sKP + 256 * 64;    // B for output      [c][r]
  unsigned short* sP  = sVP + 64 * 256;    // probs, A for out  [l][r]

  const int bh = blockIdx.x >> 6;  // 64 L-tiles per (b,h)
  const int l0 = (blockIdx.x & 63) * 64;
  const int b  = bh >> 4;
  const int h  = bh & 15;

  const int t    = threadIdx.x;
  const int wave = t >> 5;
  const int lane = t & 31;

  {  // Async DMA of K_proj / V_proj tiles (32 KB bf16 each) into LDS.
    const char* gk = (const char*)(kp + (size_t)bh * Rr * Cc);
    const char* gv = (const char*)(vp + (size_t)bh * Cc * Rr);
    for (int i = t; i < (Rr * Cc) / 8; i += 128) {
      const unsigned lk = SKP_OFF + (unsigned)i * 16u;
      const unsigned lv = SVP_OFF + (unsigned)i * 16u;
      asm volatile("global_load_async_to_lds_b128 %0, %1, off"
                   :: "v"(lk), "v"(gk + (size_t)i * 16) : "memory");
      asm volatile("global_load_async_to_lds_b128 %0, %1, off"
                   :: "v"(lv), "v"(gv + (size_t)i * 16) : "memory");
    }
  }
  {  // Q tile: 64x64 f32 -> bf16 [l][c], packed b128 LDS stores
    const int l = t >> 1, ch = (t & 1) * 32;
    const float* src = q + (((size_t)b * Ls + l0 + l) * Hh + h) * Cc + ch;
#pragma unroll
    for (int j = 0; j < 32; j += 8) {
      v4f f0 = *reinterpret_cast<const v4f*>(src + j);
      v4f f1 = *reinterpret_cast<const v4f*>(src + j + 4);
      v4u pk4;
      pk4.x = pack_bf16x2(f0.x, f0.y);
      pk4.y = pack_bf16x2(f0.z, f0.w);
      pk4.z = pack_bf16x2(f1.x, f1.y);
      pk4.w = pack_bf16x2(f1.z, f1.w);
      *reinterpret_cast<v4u*>(&sQ[l * 64 + ch + j]) = pk4;
    }
  }
  asm volatile("s_wait_asynccnt 0x0" ::: "memory");
  __syncthreads();

  // ---- Stage 2: scores for this wave's 16-row strip, all 256 R columns ----
  const int m0 = wave * 16;
  const v8f vzero = {};
  v8f accS[16];
#pragma unroll
  for (int nt = 0; nt < 16; ++nt) accS[nt] = vzero;

#pragma unroll
  for (int kk = 0; kk < Cc; kk += 32) {
    v16bf a = load_frag_a(sQ, 64, m0, kk);
#pragma unroll
    for (int nt = 0; nt < 16; ++nt) {
      v16bf bm = load_frag_b(sKP, 64, nt * 16, kk);
      accS[nt] = __builtin_amdgcn_wmma_f32_16x16x32_bf16(
          false, a, false, bm, (short)0, accS[nt], false, false);
    }
  }

  // ---- Softmax over R=256 per row; rows e and e+8 live in the two lane halves
  const float scale = 0.125f;  // 1/sqrt(C)
  const int nlo = lane & 15;
  const int mhi = (lane >> 4) << 3;
#pragma unroll
  for (int e = 0; e < 8; ++e) {
    float x[16];
    float mx = -3.402823466e38f;
#pragma unroll
    for (int nt = 0; nt < 16; ++nt) {
      x[nt] = accS[nt][e] * scale;
      mx = fmaxf(mx, x[nt]);
    }
    mx = fmaxf(mx, __shfl_xor(mx, 8, 32));
    mx = fmaxf(mx, __shfl_xor(mx, 4, 32));
    mx = fmaxf(mx, __shfl_xor(mx, 2, 32));
    mx = fmaxf(mx, __shfl_xor(mx, 1, 32));
    float s = 0.f;
#pragma unroll
    for (int nt = 0; nt < 16; ++nt) {
      x[nt] = __expf(x[nt] - mx);
      s += x[nt];
    }
    s += __shfl_xor(s, 8, 32);
    s += __shfl_xor(s, 4, 32);
    s += __shfl_xor(s, 2, 32);
    s += __shfl_xor(s, 1, 32);
    const float inv = 1.0f / s;
    const int row = m0 + mhi + e;
#pragma unroll
    for (int nt = 0; nt < 16; ++nt)
      sP[row * 256 + nt * 16 + nlo] = f32_to_bf16u(x[nt] * inv);
  }
  __syncthreads();

  // ---- Stage 3: out = P (16 x 256) * V_proj (256 x 64) ----
  v8f accO[4];
#pragma unroll
  for (int nt = 0; nt < 4; ++nt) accO[nt] = vzero;
#pragma unroll
  for (int kr = 0; kr < Rr; kr += 32) {
    v16bf a = load_frag_a(sP, 256, m0, kr);
#pragma unroll
    for (int nt = 0; nt < 4; ++nt) {
      v16bf bm = load_frag_b(sVP, 256, nt * 16, kr);
      accO[nt] = __builtin_amdgcn_wmma_f32_16x16x32_bf16(
          false, a, false, bm, (short)0, accO[nt], false, false);
    }
  }

#pragma unroll
  for (int nt = 0; nt < 4; ++nt)
#pragma unroll
    for (int e = 0; e < 8; ++e) {
      const int l = l0 + m0 + mhi + e;
      const int c = nt * 16 + nlo;
      out[(((size_t)b * Ls + l) * Hh + h) * Cc + c] = accO[nt][e];
    }
}

// ---------------------------------------------------------------------------
extern "C" void kernel_launch(void* const* d_in, const int* in_sizes, int n_in,
                              void* d_out, int out_size, void* d_ws, size_t ws_size,
                              hipStream_t stream) {
  const float* q  = (const float*)d_in[0];
  const float* k  = (const float*)d_in[1];
  const float* v  = (const float*)d_in[2];
  const float* pk = (const float*)d_in[3];
  const float* pv = (const float*)d_in[4];

  unsigned short* kp = (unsigned short*)d_ws;                 // [B,H,R,C] bf16 (2 MB)
  unsigned short* vp = kp + (size_t)Bb * Hh * Rr * Cc;        // [B,H,C,R] bf16 (2 MB)

  linformer_project_kv<<<dim3(Bb * Hh * (Rr / 64)), dim3(128), 0, stream>>>(
      k, v, pk, pv, kp, vp);

  linformer_attn<<<dim3(Bb * Hh * (Ls / 64)), dim3(128), SMEM_BYTES, stream>>>(
      q, kp, vp, (float*)d_out);
}
